// CausalSelfAttention_67473936220514
// MI455X (gfx1250) — matrix operand order repaired and checked by reading, not compiled, not run on hardware
//
#include <hip/hip_runtime.h>
#include <hip/hip_bf16.h>

typedef __attribute__((ext_vector_type(16))) __bf16 v16bf;
typedef __attribute__((ext_vector_type(8)))  __bf16 bf16x8;
typedef __attribute__((ext_vector_type(8)))  float  v8f;

#define WMMA_BF16(A, B, C) \
    __builtin_amdgcn_wmma_f32_16x16x32_bf16(false, (A), false, (B), (short)0, (C), false, false)

// ---------------------------------------------------------------------------
// Kernel 1: f32 -> bf16 elementwise cast
// ---------------------------------------------------------------------------
__global__ __launch_bounds__(256) void cast_bf16_kernel(const float* __restrict__ src,
                                                        __bf16* __restrict__ dst, int n) {
    int i = blockIdx.x * blockDim.x + threadIdx.x;
    if (i < n) dst[i] = (__bf16)src[i];
}

// ---------------------------------------------------------------------------
// Kernel 2: W [K x N] f32 row-major -> Wt [N x K] bf16 (column-major of W)
// ---------------------------------------------------------------------------
__global__ __launch_bounds__(256) void cast_transpose_kernel(const float* __restrict__ W,
                                                             __bf16* __restrict__ Wt,
                                                             int K, int N) {
    int i = blockIdx.x * blockDim.x + threadIdx.x;
    if (i < K * N) {
        int k = i / N, n = i % N;
        Wt[(size_t)n * K + k] = (__bf16)W[i];
    }
}

// ---------------------------------------------------------------------------
// Fragment load helpers (per ISA 7.12.2 layouts, wave32)
// ---------------------------------------------------------------------------
// A-matrix 16x32 bf16: lane = (m = lane&15, khalf = lane>>4)
//   elems 0..7  = K[8*khalf  .. 8*khalf+8)
//   elems 8..15 = K[16+8*khalf .. +8)
__device__ inline v16bf load_a_frag(const __bf16* rowptr, int khalf) {
    v16bf a;
    *(bf16x8*)&a       = *(const bf16x8*)(rowptr + 8 * khalf);
    *((bf16x8*)&a + 1) = *(const bf16x8*)(rowptr + 16 + 8 * khalf);
    return a;
}
// B-matrix 32x16 bf16 from column-major storage: lane = (n = lane&15, khalf)
//   elems 0..7 = K[8*khalf .. +8), elems 8..15 = K[16+8*khalf .. +8) for column n
__device__ inline v16bf load_b_frag(const __bf16* colptr, int khalf) {
    v16bf b;
    *(bf16x8*)&b       = *(const bf16x8*)(colptr + 8 * khalf);
    *((bf16x8*)&b + 1) = *(const bf16x8*)(colptr + 16 + 8 * khalf);
    return b;
}

// ---------------------------------------------------------------------------
// Kernel 3: QKV projection GEMM
//   A  = x_bf16   [M=8192, K=1024] row-major
//   Bt = Wqkv^T   [N=3072, K=1024] bf16 (column-major of W)
//   out: n in [0,2048)   -> qk[m, n]      (Q then K, row-major bf16)
//        n in [2048,3072)-> vt[bh][d][t]  (V transposed for P*V B-frags)
// Block = 128 threads (4 waves, 2x2), block tile 128x128, wave tile 64x64.
// 16 WMMAs per 8 fragment loads per K-step (4x reuse of A and B).
// ---------------------------------------------------------------------------
__global__ __launch_bounds__(128) void gemm_qkv_kernel(const __bf16* __restrict__ A,
                                                       const __bf16* __restrict__ Bt,
                                                       const float* __restrict__ bias,
                                                       __bf16* __restrict__ qk,
                                                       __bf16* __restrict__ vt) {
    const int K = 1024;
    const int lane = threadIdx.x & 31;
    const int wave = threadIdx.x >> 5;
    const int m0 = blockIdx.y * 128 + (wave >> 1) * 64;
    const int n0 = blockIdx.x * 128 + (wave & 1) * 64;
    const int mrow = lane & 15, khalf = lane >> 4;

    v8f acc[4][4] = {};
    for (int k = 0; k < K; k += 32) {
        v16bf a[4], b[4];
        #pragma unroll
        for (int i = 0; i < 4; i++) {
            const __bf16* ap = A + (size_t)(m0 + i * 16 + mrow) * K + k;
            a[i] = load_a_frag(ap, khalf);
        }
        #pragma unroll
        for (int j = 0; j < 4; j++) {
            const __bf16* bp = Bt + (size_t)(n0 + j * 16 + mrow) * K + k;
            b[j] = load_b_frag(bp, khalf);
        }
        // pull the next K-step into the near caches (global_prefetch_b8)
        if (k + 32 < K) {
            __builtin_prefetch(A + (size_t)(m0 + mrow) * K + k + 32, 0, 0);
            __builtin_prefetch(Bt + (size_t)(n0 + mrow) * K + k + 32, 0, 0);
        }
        #pragma unroll
        for (int i = 0; i < 4; i++)
            #pragma unroll
            for (int j = 0; j < 4; j++)
                acc[i][j] = WMMA_BF16(a[i], b[j], acc[i][j]);
    }

    #pragma unroll
    for (int j = 0; j < 4; j++) {
        // uniform per 16-wide tile: force a scalar branch (no exec-mask split)
        const bool is_qk = __builtin_amdgcn_readfirstlane(n0 + j * 16) < 2048;
        const int ncol = n0 + j * 16 + mrow;
        const float bv = bias[ncol];
        #pragma unroll
        for (int i = 0; i < 4; i++) {
            #pragma unroll
            for (int r = 0; r < 8; r++) {
                const int m = m0 + i * 16 + r + 8 * khalf;
                const float v = acc[i][j][r] + bv;
                if (is_qk) {                             // Q or K, row-major
                    qk[(size_t)m * 2048 + ncol] = (__bf16)v;
                } else {                                 // V -> transposed
                    const int np = ncol - 2048;
                    const int h = np >> 6, d = np & 63;
                    const int bb = m >> 11, t = m & 2047;
                    vt[(((size_t)(bb * 16 + h)) * 64 + d) * 2048 + t] = (__bf16)v;
                }
            }
        }
    }
}

// ---------------------------------------------------------------------------
// Kernel 4: flash attention, one wave per 16-query tile (B=4,H=16,T=2048,Dh=64)
// ---------------------------------------------------------------------------
__global__ __launch_bounds__(128) void attn_kernel(const __bf16* __restrict__ qk,
                                                   const __bf16* __restrict__ vt,
                                                   __bf16* __restrict__ ctx) {
    __shared__ __attribute__((aligned(16))) __bf16 plds[4][16 * 32];
    const int lane = threadIdx.x & 31;
    const int wave = threadIdx.x >> 5;
    const int idx  = blockIdx.x * 4 + wave;       // 8192 waves total
    const int qt = idx & 127;                     // query tile within (b,h)
    const int bh = idx >> 7;
    const int h = bh & 15, b = bh >> 4;
    const int T = 2048, C2 = 2048;
    const int qbase = qt * 16;
    const int mrow = lane & 15, khalf = lane >> 4;

    const __bf16* qptr = qk + (size_t)b * T * C2 + h * 64;
    const __bf16* kptr = qptr + 1024;
    const __bf16* vptr = vt + (size_t)bh * 64 * 2048;   // [d][t]

    // Q fragments for the two 32-wide head-dim chunks, held for the whole loop
    v16bf qa[2];
    {
        const __bf16* qp = qptr + (size_t)(qbase + mrow) * C2;
        qa[0] = load_a_frag(qp, khalf);
        qa[1] = load_a_frag(qp + 32, khalf);
    }

    float mrw[8], lrw[8];
    v8f o[4] = {};
    #pragma unroll
    for (int r = 0; r < 8; r++) { mrw[r] = -1e30f; lrw[r] = 0.0f; }
    const float scale = 0.125f;   // 1/sqrt(64)

    const int kend = qbase + 16;  // keys strictly <= last query row
    for (int kb = 0; kb < kend; kb += 32) {
        // ---- S = Q K^T for 32 keys: two 16x16 tiles, 2 WMMAs each ----
        v8f s[2] = {};
        #pragma unroll
        for (int c = 0; c < 2; c++) {
            const __bf16* kp = kptr + (size_t)(kb + c * 16 + mrow) * C2;
            #pragma unroll
            for (int kk = 0; kk < 2; kk++) {
                v16bf bf = load_b_frag(kp + kk * 32, khalf);
                s[c] = WMMA_BF16(qa[kk], bf, s[c]);
            }
        }
        // ---- scale + causal mask ----
        #pragma unroll
        for (int c = 0; c < 2; c++) {
            const int kj = kb + c * 16 + mrow;
            #pragma unroll
            for (int r = 0; r < 8; r++) {
                const int qi = qbase + r + 8 * khalf;
                float sv = s[c][r] * scale;
                s[c][r] = (kj > qi) ? -1e30f : sv;
            }
        }
        // ---- online softmax (row = vgpr r within 16-lane group) ----
        float bm[8], rs[8], alpha[8];
        #pragma unroll
        for (int r = 0; r < 8; r++) bm[r] = fmaxf(s[0][r], s[1][r]);
        #pragma unroll
        for (int off = 1; off < 16; off <<= 1)
            #pragma unroll
            for (int r = 0; r < 8; r++)
                bm[r] = fmaxf(bm[r], __shfl_xor(bm[r], off, 32));
        #pragma unroll
        for (int r = 0; r < 8; r++) {
            const float nm = fmaxf(mrw[r], bm[r]);
            alpha[r] = __expf(mrw[r] - nm);
            mrw[r] = nm;
            rs[r] = 0.0f;
        }
        #pragma unroll
        for (int c = 0; c < 2; c++)
            #pragma unroll
            for (int r = 0; r < 8; r++) {
                const float p = __expf(s[c][r] - mrw[r]);
                s[c][r] = p;
                rs[r] += p;
            }
        #pragma unroll
        for (int off = 1; off < 16; off <<= 1)
            #pragma unroll
            for (int r = 0; r < 8; r++) rs[r] += __shfl_xor(rs[r], off, 32);
        #pragma unroll
        for (int r = 0; r < 8; r++) lrw[r] = lrw[r] * alpha[r] + rs[r];
        #pragma unroll
        for (int j = 0; j < 4; j++)
            #pragma unroll
            for (int r = 0; r < 8; r++) o[j][r] *= alpha[r];

        // ---- P (C-layout f32) -> bf16 A-fragment via LDS bounce ----
        __bf16* pl = &plds[wave][0];
        #pragma unroll
        for (int c = 0; c < 2; c++)
            #pragma unroll
            for (int r = 0; r < 8; r++) {
                const int m = r + 8 * khalf;
                pl[m * 32 + c * 16 + mrow] = (__bf16)s[c][r];
            }
        asm volatile("s_wait_dscnt 0" ::: "memory");   // intra-wave LDS RAW
        v16bf pa = load_a_frag(pl + mrow * 32, khalf);

        // ---- O += P @ V : 4 dim-tiles of 16 ----
        #pragma unroll
        for (int j = 0; j < 4; j++) {
            v16bf vb = load_b_frag(vptr + (size_t)(j * 16 + mrow) * 2048 + kb, khalf);
            o[j] = WMMA_BF16(pa, vb, o[j]);
        }
    }

    // ---- finalize: divide by row sums, write ctx [8192, 1024] bf16 ----
    __bf16* cp = ctx + (size_t)b * T * 1024 + h * 64;
    #pragma unroll
    for (int j = 0; j < 4; j++)
        #pragma unroll
        for (int r = 0; r < 8; r++) {
            const int m = qbase + r + 8 * khalf;
            cp[(size_t)m * 1024 + j * 16 + mrow] = (__bf16)(o[j][r] / lrw[r]);
        }
}

// ---------------------------------------------------------------------------
// Kernel 5: output projection GEMM, f32 result
//   A = ctx bf16 [8192,1024], Bt = Wout^T bf16 [1024,1024], out f32 [8192,1024]
// ---------------------------------------------------------------------------
__global__ __launch_bounds__(128) void gemm_out_kernel(const __bf16* __restrict__ A,
                                                       const __bf16* __restrict__ Bt,
                                                       const float* __restrict__ bias,
                                                       float* __restrict__ out) {
    const int K = 1024;
    const int lane = threadIdx.x & 31;
    const int wave = threadIdx.x >> 5;
    const int m0 = blockIdx.y * 128 + (wave >> 1) * 64;
    const int n0 = blockIdx.x * 128 + (wave & 1) * 64;
    const int mrow = lane & 15, khalf = lane >> 4;

    v8f acc[4][4] = {};
    for (int k = 0; k < K; k += 32) {
        v16bf a[4], b[4];
        #pragma unroll
        for (int i = 0; i < 4; i++)
            a[i] = load_a_frag(A + (size_t)(m0 + i * 16 + mrow) * K + k, khalf);
        #pragma unroll
        for (int j = 0; j < 4; j++)
            b[j] = load_b_frag(Bt + (size_t)(n0 + j * 16 + mrow) * K + k, khalf);
        if (k + 32 < K) {
            __builtin_prefetch(A + (size_t)(m0 + mrow) * K + k + 32, 0, 0);
            __builtin_prefetch(Bt + (size_t)(n0 + mrow) * K + k + 32, 0, 0);
        }
        #pragma unroll
        for (int i = 0; i < 4; i++)
            #pragma unroll
            for (int j = 0; j < 4; j++)
                acc[i][j] = WMMA_BF16(a[i], b[j], acc[i][j]);
    }
    #pragma unroll
    for (int i = 0; i < 4; i++)
        #pragma unroll
        for (int j = 0; j < 4; j++) {
            const int ncol = n0 + j * 16 + mrow;
            const float bv = bias[ncol];
            #pragma unroll
            for (int r = 0; r < 8; r++) {
                const int m = m0 + i * 16 + r + 8 * khalf;
                out[(size_t)m * 1024 + ncol] = acc[i][j][r] + bv;
            }
        }
}

// ---------------------------------------------------------------------------
// Host launcher
// ---------------------------------------------------------------------------
extern "C" void kernel_launch(void* const* d_in, const int* in_sizes, int n_in,
                              void* d_out, int out_size, void* d_ws, size_t ws_size,
                              hipStream_t stream) {
    const float* x     = (const float*)d_in[0];   // [4,2048,1024]
    const float* W_qkv = (const float*)d_in[1];   // [1024,3072]
    const float* b_qkv = (const float*)d_in[2];   // [3072]
    const float* W_out = (const float*)d_in[3];   // [1024,1024]
    const float* b_out = (const float*)d_in[4];   // [1024]
    float* out = (float*)d_out;                   // [8192,1024] f32

    const int M = 8192, C = 1024;

    size_t off = 0;
    auto carve = [&](size_t bytes) -> char* {
        char* p = (char*)d_ws + off;
        off += (bytes + 255) & ~(size_t)255;
        return p;
    };
    __bf16* xb    = (__bf16*)carve((size_t)M * C * 2);          // 16 MB
    __bf16* wqkvT = (__bf16*)carve((size_t)3072 * 1024 * 2);    // 6 MB
    __bf16* woutT = (__bf16*)carve((size_t)1024 * 1024 * 2);    // 2 MB
    __bf16* qkbuf = (__bf16*)carve((size_t)M * 2048 * 2);       // 32 MB
    __bf16* vtbuf = (__bf16*)carve((size_t)64 * 64 * 2048 * 2); // 16 MB
    __bf16* ctx   = (__bf16*)carve((size_t)M * C * 2);          // 16 MB
    (void)ws_size; (void)in_sizes; (void)n_in; (void)out_size;

    // 1) casts
    {
        int n = M * C;
        cast_bf16_kernel<<<(n + 255) / 256, 256, 0, stream>>>(x, xb, n);
    }
    {
        int n = 1024 * 3072;
        cast_transpose_kernel<<<(n + 255) / 256, 256, 0, stream>>>(W_qkv, wqkvT, 1024, 3072);
    }
    {
        int n = 1024 * 1024;
        cast_transpose_kernel<<<(n + 255) / 256, 256, 0, stream>>>(W_out, woutT, 1024, 1024);
    }
    // 2) QKV projection: grid (N/128, M/128)
    gemm_qkv_kernel<<<dim3(3072 / 128, M / 128), 128, 0, stream>>>(xb, wqkvT, b_qkv, qkbuf, vtbuf);
    // 3) flash attention: 8192 waves / 4 per block
    attn_kernel<<<2048, 128, 0, stream>>>(qkbuf, vtbuf, ctx);
    // 4) output projection
    gemm_out_kernel<<<dim3(1024 / 128, M / 128), 128, 0, stream>>>(ctx, woutT, b_out, out);
}
